// Jha_GAT_1898375544918
// MI455X (gfx1250) — matrix operand order repaired and checked by reading, not compiled, not run on hardware
//
#include <hip/hip_runtime.h>
#include <hip/hip_bf16.h>

// ---------------------------------------------------------------------------
// Problem constants (from reference)
// ---------------------------------------------------------------------------
constexpr int NN = 50000;    // nodes per channel
constexpr int EE = 800000;   // edges per channel
constexpr int EN = EE + NN;  // edges + self loops
constexpr int FF = 128;      // input feature dim
constexpr int DD = 128;      // output dim (H*C)
constexpr int HH = 4;        // heads
constexpr int BB = 64;       // graphs in batch

typedef __bf16 v16bf __attribute__((ext_vector_type(16)));
typedef float  v8f   __attribute__((ext_vector_type(8)));

__device__ __forceinline__ float lrelu01(float v) { return v > 0.f ? v : 0.01f * v; }

// ---------------------------------------------------------------------------
// Fused dual WMMA GEMM:
//   YL[M x 128] = X @ Wl + bl ;  YR[M x 128] = X @ Wr + br
// A-fragments (rows of X) are loaded ONCE and fed to two WMMAs -> halves the
// HBM traffic of the memory-bound transform phase. Both weight matrices are
// staged per-workgroup in LDS, bf16-converted and pre-swizzled into WMMA
// B-fragment order so each fragment is one aligned 32-byte LDS read.
// A and B fragments use contiguous k order (element e = k 16*half + e), not the ISA table order; f32 C/D per ISA.
// ---------------------------------------------------------------------------
__global__ __launch_bounds__(256) void gemm128_dual_wmma(
    const float* __restrict__ X,
    const float* __restrict__ Wl, const float* __restrict__ bl,
    const float* __restrict__ Wr, const float* __restrict__ br,
    float* __restrict__ YL, float* __restrict__ YR, int M)
{
    // LDS: 2 weight matrices in fragment order:
    //   sW[m][ ((kchunk*128 + col)*32) + half*16 + e ]   (bf16)
    __shared__ __align__(32) __bf16 sW[2][FF * DD];

    // ---- stage Wl / Wr into LDS (float4 global reads, bf16 scatter stores)
    for (int i = threadIdx.x; i < (FF * DD) / 4; i += 256) {
        int base = i * 4;                 // row-major [k][n], 4 consecutive n
        int k = base >> 7, n = base & 127;
        int kc = k >> 5, kl = k & 31;
        int d0 = ((kc * 128 + n) * 32) + (kl >> 4) * 16 + (kl & 15);
        float4 vl = ((const float4*)Wl)[i];
        float4 vr = ((const float4*)Wr)[i];
        sW[0][d0]      = (__bf16)vl.x; sW[0][d0 + 32] = (__bf16)vl.y;
        sW[0][d0 + 64] = (__bf16)vl.z; sW[0][d0 + 96] = (__bf16)vl.w;
        sW[1][d0]      = (__bf16)vr.x; sW[1][d0 + 32] = (__bf16)vr.y;
        sW[1][d0 + 64] = (__bf16)vr.z; sW[1][d0 + 96] = (__bf16)vr.w;
    }
    __syncthreads();

    int wave = (blockIdx.x * blockDim.x + threadIdx.x) >> 5;
    int lane = threadIdx.x & 31;
    int tm = wave >> 3;        // D/16 = 8 column tiles per row tile
    int tn = wave & 7;
    if (tm >= M / 16) return;  // wave-uniform guard (EXEC stays all-ones)

    int half = lane >> 4;      // 0: lanes 0-15, 1: lanes 16-31
    int m    = lane & 15;
    int rowA = tm * 16 + m;    // A-matrix row (both halves hold same rows)
    int col  = tn * 16 + m;    // B/C/D column (N = lane & 15)

    v8f accL = {};
    v8f accR = {};
    #pragma unroll
    for (int kk = 0; kk < FF; kk += 32) {
        // ---- A fragment: lane holds K = {half*16 .. +15}
        const float* xr = X + (size_t)rowA * FF + kk;
        float4 f0 = *(const float4*)(xr + half * 16);
        float4 f1 = *(const float4*)(xr + half * 16 + 4);
        float4 f2 = *(const float4*)(xr + half * 16 + 8);
        float4 f3 = *(const float4*)(xr + half * 16 + 12);
        v16bf a;
        a[0]  = (__bf16)f0.x; a[1]  = (__bf16)f0.y; a[2]  = (__bf16)f0.z; a[3]  = (__bf16)f0.w;
        a[4]  = (__bf16)f1.x; a[5]  = (__bf16)f1.y; a[6]  = (__bf16)f1.z; a[7]  = (__bf16)f1.w;
        a[8]  = (__bf16)f2.x; a[9]  = (__bf16)f2.y; a[10] = (__bf16)f2.z; a[11] = (__bf16)f2.w;
        a[12] = (__bf16)f3.x; a[13] = (__bf16)f3.y; a[14] = (__bf16)f3.z; a[15] = (__bf16)f3.w;

        // ---- B fragments: single aligned 32B LDS read each
        int bidx = (((kk >> 5) * 128 + col) * 32) + half * 16;
        v16bf bL = *(const v16bf*)&sW[0][bidx];
        v16bf bR = *(const v16bf*)&sW[1][bidx];

        accL = __builtin_amdgcn_wmma_f32_16x16x32_bf16(
                   false, a, false, bL, (short)0, accL, false, false);
        accR = __builtin_amdgcn_wmma_f32_16x16x32_bf16(
                   false, a, false, bR, (short)0, accR, false, false);
    }

    float bvL = bl[col];
    float bvR = br[col];
    #pragma unroll
    for (int r = 0; r < 8; ++r) {       // VGPR r -> row M = r + 8*half
        int row = tm * 16 + r + half * 8;
        YL[(size_t)row * DD + col] = accL[r] + bvL;
        YR[(size_t)row * DD + col] = accR[r] + bvR;
    }
}

// ---------------------------------------------------------------------------
// Buffer init kernels
// ---------------------------------------------------------------------------
__global__ void set_f32(float* p, int n, float v)
{
    int i = blockIdx.x * blockDim.x + threadIdx.x;
    if (i < n) p[i] = v;
}
__global__ void set_i32(int* p, int n, int v)
{
    int i = blockIdx.x * blockDim.x + threadIdx.x;
    if (i < n) p[i] = v;
}

// order-preserving float -> int key for atomicMax
__device__ __forceinline__ int fkey(float f)
{
    int i = __float_as_int(f);
    return i >= 0 ? i : (i ^ 0x7fffffff);
}
__device__ __forceinline__ float funkey(int k)
{
    return k >= 0 ? __int_as_float(k) : __int_as_float(k ^ 0x7fffffff);
}

// ---------------------------------------------------------------------------
// Pass 1: per-edge attention logits + per-(dst,head) running max.
// One wave32 per edge: lane l handles channel c=l of every head (idx = h*32+l).
// ---------------------------------------------------------------------------
__global__ __launch_bounds__(256) void edge_logits(
    const float* __restrict__ XL, const float* __restrict__ XR,
    const int* __restrict__ ei, const float* __restrict__ att,
    float* __restrict__ ELOG, int* __restrict__ EMAX)
{
    int gw   = (blockIdx.x * 256 + threadIdx.x) >> 5;
    int lane = threadIdx.x & 31;
    if (gw >= EN) return;

    int src = gw < EE ? ei[gw]      : gw - EE;
    int dst = gw < EE ? ei[EE + gw] : gw - EE;

    const float* xl = XL + (size_t)src * DD;
    const float* xr = XR + (size_t)dst * DD;

    float p0, p1, p2, p3;
    {
        float s;
        s = xl[lane]       + xr[lane];       s = s > 0.f ? s : 0.2f * s; p0 = s * att[lane];
        s = xl[32 + lane]  + xr[32 + lane];  s = s > 0.f ? s : 0.2f * s; p1 = s * att[32 + lane];
        s = xl[64 + lane]  + xr[64 + lane];  s = s > 0.f ? s : 0.2f * s; p2 = s * att[64 + lane];
        s = xl[96 + lane]  + xr[96 + lane];  s = s > 0.f ? s : 0.2f * s; p3 = s * att[96 + lane];
    }
    #pragma unroll
    for (int off = 16; off; off >>= 1) {
        p0 += __shfl_xor(p0, off, 32);
        p1 += __shfl_xor(p1, off, 32);
        p2 += __shfl_xor(p2, off, 32);
        p3 += __shfl_xor(p3, off, 32);
    }
    if (lane < 4) {
        float v = lane == 0 ? p0 : lane == 1 ? p1 : lane == 2 ? p2 : p3;
        ELOG[(size_t)gw * HH + lane] = v;
        atomicMax(EMAX + (size_t)dst * HH + lane, fkey(v));
    }
}

// ---------------------------------------------------------------------------
// Pass 2: ee = exp(e - max); accumulate denom[dst,h] and numer[dst,:] = ee*xl[src]
// ---------------------------------------------------------------------------
__global__ __launch_bounds__(256) void edge_accum(
    const float* __restrict__ XL, const int* __restrict__ ei,
    const float* __restrict__ ELOG, const int* __restrict__ EMAX,
    float* __restrict__ DEN, float* __restrict__ ACC)
{
    int gw   = (blockIdx.x * 256 + threadIdx.x) >> 5;
    int lane = threadIdx.x & 31;
    if (gw >= EN) return;

    int src = gw < EE ? ei[gw]      : gw - EE;
    int dst = gw < EE ? ei[EE + gw] : gw - EE;

    float ee = 0.f;
    if (lane < 4) {
        float e = ELOG[(size_t)gw * HH + lane];
        float m = funkey(EMAX[(size_t)dst * HH + lane]);
        ee = __expf(e - m);
        atomicAdd(DEN + (size_t)dst * HH + lane, ee);
    }
    float e0 = __shfl(ee, 0, 32);
    float e1 = __shfl(ee, 1, 32);
    float e2 = __shfl(ee, 2, 32);
    float e3 = __shfl(ee, 3, 32);

    const float* xl = XL + (size_t)src * DD;
    float* ap       = ACC + (size_t)dst * DD;
    atomicAdd(ap + lane,       e0 * xl[lane]);
    atomicAdd(ap + 32 + lane,  e1 * xl[32 + lane]);
    atomicAdd(ap + 64 + lane,  e2 * xl[64 + lane]);
    atomicAdd(ap + 96 + lane,  e3 * xl[96 + lane]);
}

// ---------------------------------------------------------------------------
// Node finalize: out = lrelu(numer/denom + bias); pool into per-graph sums.
// One wave32 per node.
// ---------------------------------------------------------------------------
__global__ __launch_bounds__(256) void node_fin(
    const float* __restrict__ ACC, const float* __restrict__ DEN,
    const float* __restrict__ bias, const int* __restrict__ batch,
    float* __restrict__ POOL, float* __restrict__ CNT)
{
    int gw   = (blockIdx.x * 256 + threadIdx.x) >> 5;
    int lane = threadIdx.x & 31;
    if (gw >= NN) return;

    int g = batch[gw];
    const float* ap = ACC + (size_t)gw * DD;
    float* pp       = POOL + (size_t)g * DD;
    #pragma unroll
    for (int h = 0; h < HH; ++h) {
        int idx = h * 32 + lane;
        float d = fmaxf(DEN[(size_t)gw * HH + h], 1e-16f);
        float v = lrelu01(ap[idx] / d + bias[idx]);
        atomicAdd(pp + idx, v);
    }
    if (lane == 0) atomicAdd(CNT + g, 1.0f);
}

// ---------------------------------------------------------------------------
// Per-branch FC: FCO[g,j] = lrelu( (POOL[g,:]/cnt) @ fcW + fcb )
// ---------------------------------------------------------------------------
__global__ __launch_bounds__(256) void branch_fc(
    const float* __restrict__ POOL, const float* __restrict__ CNT,
    const float* __restrict__ Wfc, const float* __restrict__ bfc,
    float* __restrict__ FCO)
{
    int t = blockIdx.x * blockDim.x + threadIdx.x;
    if (t >= BB * DD) return;
    int g = t >> 7, j = t & 127;
    const float* pv = POOL + (size_t)g * DD;
    float acc = 0.f;
    for (int k = 0; k < DD; ++k) acc += pv[k] * Wfc[(size_t)k * DD + j];
    float c = fmaxf(CNT[g], 1.0f);
    FCO[t] = lrelu01(acc / c + bfc[j]);
}

// ---------------------------------------------------------------------------
// Head: concat -> fc1(256->128) -> lrelu -> fc2(128->16) -> lrelu -> out(16->1)
// One block of 128 threads per graph.
// ---------------------------------------------------------------------------
__global__ __launch_bounds__(128) void head_mlp(
    const float* __restrict__ FCO1, const float* __restrict__ FCO2,
    const float* __restrict__ W1, const float* __restrict__ b1,
    const float* __restrict__ W2, const float* __restrict__ b2,
    const float* __restrict__ Wo, const float* __restrict__ bo,
    float* __restrict__ out)
{
    __shared__ float h1[128];
    __shared__ float h2[16];
    int g = blockIdx.x, j = threadIdx.x;
    const float* a = FCO1 + (size_t)g * DD;
    const float* b = FCO2 + (size_t)g * DD;

    float acc = b1[j];
    for (int k = 0; k < 128; ++k) acc += a[k] * W1[(size_t)k * 128 + j];
    for (int k = 0; k < 128; ++k) acc += b[k] * W1[(size_t)(128 + k) * 128 + j];
    h1[j] = lrelu01(acc);
    __syncthreads();

    if (j < 16) {
        float s = b2[j];
        for (int k = 0; k < 128; ++k) s += h1[k] * W2[(size_t)k * 16 + j];
        h2[j] = lrelu01(s);
    }
    __syncthreads();

    if (j == 0) {
        float s = bo[0];
        for (int k = 0; k < 16; ++k) s += h2[k] * Wo[k];
        out[g] = s;
    }
}

// ---------------------------------------------------------------------------
// Host-side orchestration
// ---------------------------------------------------------------------------
extern "C" void kernel_launch(void* const* d_in, const int* in_sizes, int n_in,
                              void* d_out, int out_size, void* d_ws, size_t ws_size,
                              hipStream_t stream)
{
    (void)in_sizes; (void)n_in; (void)out_size; (void)ws_size;

    const float* ch_x[2]     = { (const float*)d_in[0], (const float*)d_in[3] };
    const int*   ch_ei[2]    = { (const int*)d_in[1],   (const int*)d_in[4]   };
    const int*   ch_batch[2] = { (const int*)d_in[2],   (const int*)d_in[5]   };

    // p1 at 6..13, p2 at 14..21 : Wl, bl, Wr, br, att, bias, fcW, fcb
    const float* Wl[2]   = { (const float*)d_in[6],  (const float*)d_in[14] };
    const float* bl[2]   = { (const float*)d_in[7],  (const float*)d_in[15] };
    const float* Wr[2]   = { (const float*)d_in[8],  (const float*)d_in[16] };
    const float* br[2]   = { (const float*)d_in[9],  (const float*)d_in[17] };
    const float* att[2]  = { (const float*)d_in[10], (const float*)d_in[18] };
    const float* bias[2] = { (const float*)d_in[11], (const float*)d_in[19] };
    const float* fcW[2]  = { (const float*)d_in[12], (const float*)d_in[20] };
    const float* fcb[2]  = { (const float*)d_in[13], (const float*)d_in[21] };

    const float* fc1_W = (const float*)d_in[22];
    const float* fc1_b = (const float*)d_in[23];
    const float* fc2_W = (const float*)d_in[24];
    const float* fc2_b = (const float*)d_in[25];
    const float* out_W = (const float*)d_in[26];
    const float* out_b = (const float*)d_in[27];

    char* ws = (char*)d_ws;
    size_t off = 0;
    auto take = [&](size_t bytes) -> void* {
        void* p = ws + off;
        off += (bytes + 255) & ~(size_t)255;
        return p;
    };

    float* FCO[2];
    for (int c = 0; c < 2; ++c) {
        float* XL   = (float*)take((size_t)NN * DD * 4);
        float* XR   = (float*)take((size_t)NN * DD * 4);   // reused as ACC after pass 1
        float* ELOG = (float*)take((size_t)EN * HH * 4);
        int*   EMAX = (int*)  take((size_t)NN * HH * 4);
        float* DEN  = (float*)take((size_t)NN * HH * 4);
        float* POOL = (float*)take((size_t)BB * DD * 4);
        float* CNT  = (float*)take((size_t)BB * 4);
        FCO[c]      = (float*)take((size_t)BB * DD * 4);
        float* ACC  = XR;

        // 1) fused linear transforms (WMMA, X read once for both Wl and Wr)
        dim3 gg((NN / 16) * (DD / 16) / 8);   // one wave per 16x16 output tile
        gemm128_dual_wmma<<<gg, 256, 0, stream>>>(ch_x[c], Wl[c], bl[c],
                                                  Wr[c], br[c], XL, XR, NN);

        // 2) init per-node state
        set_i32<<<(NN * HH + 255) / 256, 256, 0, stream>>>(EMAX, NN * HH, (int)0x80000000);
        set_f32<<<(NN * HH + 255) / 256, 256, 0, stream>>>(DEN, NN * HH, 0.f);
        set_f32<<<(BB * DD + 255) / 256, 256, 0, stream>>>(POOL, BB * DD, 0.f);
        set_f32<<<(BB + 255) / 256, 256, 0, stream>>>(CNT, BB, 0.f);

        // 3) edge pass 1: logits + segment max (wave per edge)
        dim3 ge(EN / 8);  // EN*32 threads / 256
        edge_logits<<<ge, 256, 0, stream>>>(XL, XR, ch_ei[c], att[c], ELOG, EMAX);

        // 4) XR dead -> zero and reuse as accumulator
        set_f32<<<((NN * DD) + 255) / 256, 256, 0, stream>>>(ACC, NN * DD, 0.f);

        // 5) edge pass 2: exp / denom / weighted scatter-add
        edge_accum<<<ge, 256, 0, stream>>>(XL, ch_ei[c], ELOG, EMAX, DEN, ACC);

        // 6) node finalize + mean-pool accumulation (wave per node)
        node_fin<<<NN / 8, 256, 0, stream>>>(ACC, DEN, bias[c], ch_batch[c], POOL, CNT);

        // 7) per-branch FC
        branch_fc<<<(BB * DD + 255) / 256, 256, 0, stream>>>(POOL, CNT, fcW[c], fcb[c], FCO[c]);
    }

    // 8) head MLP
    head_mlp<<<BB, 128, 0, stream>>>(FCO[0], FCO[1], fc1_W, fc1_b, fc2_W, fc2_b,
                                     out_W, out_b, (float*)d_out);
}
